// Head_44521630990986
// MI455X (gfx1250) — matrix operand order repaired and checked by reading, not compiled, not run on hardware
//
#include <hip/hip_runtime.h>
#include <hip/hip_bf16.h>

typedef __attribute__((ext_vector_type(16))) _Float16 v16h;
typedef __attribute__((ext_vector_type(8)))  _Float16 v8h;
typedef __attribute__((ext_vector_type(8)))  float    v8f;

#define T_SEQ 256
#define C_DIM 64
#define H_DIM 64
#define SQK 72     // q/k row stride (f16 elems): 144B, 16B-aligned, conflict-free
#define SVT 264    // vT row stride: 528B == 16 mod 256 -> conflict-free b128 access
#define SWT 72     // Wt row stride

// LDS byte offsets (all 16B aligned)
#define OFF_QH   0                     // 256*72*2 = 36864
#define OFF_KH   36864                 // 36864
#define OFF_VT   73728                 // 64*264*2 = 33792
#define OFF_WT   107520                // 3*64*72*2 = 27648
#define OFF_B    135168                // 3*64*4 = 768
#define OFF_PS   107520                // P scratch overlays Wt after phase 2
#define SMEM_BYTES 135936

__device__ __forceinline__ v8f wmma16(v16h a, v16h b, v8f c) {
  return __builtin_amdgcn_wmma_f32_16x16x32_f16(false, a, false, b, (short)0, c,
                                                false, false);
}

// A operand 16x32 f16 from LDS. lane l = row; hi selects K group.
// VGPR0-3: K = k0+8*hi+0..7 ; VGPR4-7: K = k0+8*hi+16..23  (two b128 loads)
__device__ __forceinline__ v16h loadA(const _Float16* base, int stride, int l,
                                      int hi, int k0) {
  v16h a;
  const _Float16* p = base + l * stride + k0 + hi * 8;
#pragma unroll
  for (int t = 0; t < 8; ++t) { a[t] = p[t]; a[8 + t] = p[16 + t]; }
  return a;
}

// A operand 16x32 f16 built from global f32 rows (four global_load_b128 + cvt).
__device__ __forceinline__ v16h loadAg(const float* base, int stride, int l,
                                       int hi, int k0) {
  v16h a;
  const float* p = base + l * stride + k0 + hi * 8;
#pragma unroll
  for (int t = 0; t < 8; ++t) { a[t] = (_Float16)p[t]; a[8 + t] = (_Float16)p[16 + t]; }
  return a;
}

// B operand 32x16 f16. lane n = column N; VGPR0-3: K = k0+8*hi+0..7 ;
// VGPR4-7: K = k0+8*hi+16..23. Source: base[n*stride + K]  (two b128 loads)
__device__ __forceinline__ v16h loadB(const _Float16* base, int stride, int n,
                                      int hi, int k0) {
  v16h b;
  const _Float16* p = base + n * stride + k0 + hi * 8;
#pragma unroll
  for (int t = 0; t < 8; ++t) { b[t] = p[t]; b[8 + t] = p[16 + t]; }
  return b;
}

__global__ __launch_bounds__(256, 1)
void head_attn_kernel(const float* __restrict__ x,
                      const float* __restrict__ Wk, const float* __restrict__ bk,
                      const float* __restrict__ Wq, const float* __restrict__ bq,
                      const float* __restrict__ Wv, const float* __restrict__ bv,
                      float* __restrict__ out) {
  __shared__ __align__(16) unsigned char smem[SMEM_BYTES];
  _Float16* qh = (_Float16*)(smem + OFF_QH);   // q[row][h]
  _Float16* kh = (_Float16*)(smem + OFF_KH);   // k[row][h]
  _Float16* vT = (_Float16*)(smem + OFF_VT);   // v transposed: vT[h][row]
  _Float16* Wt = (_Float16*)(smem + OFF_WT);   // W^T: Wt[mat][h*SWT + c]
  float*    bb = (float*)   (smem + OFF_B);    // biases [3][64]
  _Float16* Ps = (_Float16*)(smem + OFF_PS);   // per-wave P scratch 16x32

  const int tid  = threadIdx.x;
  const int lane = tid & 31;
  const int wvu  = __builtin_amdgcn_readfirstlane(tid >> 5);  // scalar wave id
  const int l    = lane & 15;
  const int hi   = lane >> 4;
  const int b    = blockIdx.x;
  const float* xb = x + (size_t)b * T_SEQ * C_DIM;

  // ---------- Phase 1: stage W^T (f16, padded) + biases in LDS ----------
  for (int e = tid; e < 3 * 64 * 64; e += 256) {
    int m = e >> 12, rc = e & 4095, r = rc >> 6, c = rc & 63;
    const float* Wsrc = (m == 0) ? Wq : ((m == 1) ? Wk : Wv);
    Wt[m * (64 * SWT) + c * SWT + r] = (_Float16)Wsrc[rc];  // row=h, contiguous K
  }
  if (tid < 192) {
    int m = tid >> 6, c2 = tid & 63;
    const float* Bsrc = (m == 0) ? bq : ((m == 1) ? bk : bv);
    bb[tid] = Bsrc[c2];
  }
  __syncthreads();

  // ---------- Phase 2: q/k/v projections via WMMA ----------
  // 192 output tiles: rt(16) x mat(3) x ht(4); each tile = 2 WMMAs over C=64.
  for (int t2 = wvu; t2 < 192; t2 += 8) {
    int rt = t2 / 12, rem = t2 % 12, mat = rem >> 2, ht = rem & 3;   // scalars
    int r0 = rt * 16, h0 = ht * 16;
    const _Float16* wt = Wt + mat * (64 * SWT) + h0 * SWT;
    v8f acc = {};
    acc = wmma16(loadAg(xb + r0 * C_DIM, C_DIM, l, hi, 0),  loadB(wt, SWT, l, hi, 0),  acc);
    acc = wmma16(loadAg(xb + r0 * C_DIM, C_DIM, l, hi, 32), loadB(wt, SWT, l, hi, 32), acc);
    float bias = bb[mat * 64 + h0 + l];
    if (mat == 2) {                       // v -> transposed, packed b128 store
      v8h pk;
#pragma unroll
      for (int j = 0; j < 8; ++j) pk[j] = (_Float16)(acc[j] + bias);
      *(v8h*)(vT + (h0 + l) * SVT + r0 + 8 * hi) = pk;
    } else {
      _Float16* dst = (mat == 0) ? qh : kh;
#pragma unroll
      for (int j = 0; j < 8; ++j)
        dst[(r0 + j + 8 * hi) * SQK + h0 + l] = (_Float16)(acc[j] + bias);
    }
  }
  __syncthreads();

  // ---------- Phase 3: causal flash attention, 2 q-tiles per wave ----------
  _Float16* Pw = Ps + wvu * 512;          // 16x32 f16 per-wave scratch
  const float scale = 0.125f;             // 1/sqrt(64)
  // Ones B operand: column N=0 is all 1 -> O4 = P @ onesB accumulates row sums
  v16h onesB;
  {
    _Float16 ov = (l == 0) ? (_Float16)1.0f : (_Float16)0.0f;
#pragma unroll
    for (int t = 0; t < 16; ++t) onesB[t] = ov;
  }
#pragma unroll
  for (int tile2 = 0; tile2 < 2; ++tile2) {
    int i  = (tile2 == 0) ? wvu : (15 - wvu);   // scalar; pair w with 15-w
    int q0 = i * 16;
    float mrow[8];
#pragma unroll
    for (int j = 0; j < 8; ++j) mrow[j] = -__builtin_inff();
    v8f O0 = {}, O1 = {}, O2 = {}, O3 = {}, O4 = {};
    const _Float16* qbase = qh + q0 * SQK;
    v16h qa0 = loadA(qbase, SQK, l, hi, 0);
    v16h qa1 = loadA(qbase, SQK, l, hi, 32);
    int cLast = i >> 1;                   // last (diagonal) 32-key chunk; scalar
    for (int c = 0; c <= cLast; ++c) {
      int key0 = c * 32;                  // scalar
      v8f S0 = {}, S1 = {};
      const _Float16* kb = kh + key0 * SQK;
      S0 = wmma16(qa0, loadB(kb, SQK, l, hi, 0),  S0);
      S0 = wmma16(qa1, loadB(kb, SQK, l, hi, 32), S0);
      kb += 16 * SQK;
      S1 = wmma16(qa0, loadB(kb, SQK, l, hi, 0),  S1);
      S1 = wmma16(qa1, loadB(kb, SQK, l, hi, 32), S1);

      bool doMask = (c == cLast);         // scalar
#pragma unroll
      for (int j = 0; j < 8; ++j) {
        int row = q0 + j + 8 * hi;
        float s0 = S0[j] * scale;
        float s1 = S1[j] * scale;
        if (doMask) {
          s0 = (key0 + l      > row) ? -__builtin_inff() : s0;
          s1 = (key0 + 16 + l > row) ? -__builtin_inff() : s1;
        }
        // row max across the 16 lanes holding this row
        float v = fmaxf(s0, s1);
        v = fmaxf(v, __shfl_xor(v, 1, 32));
        v = fmaxf(v, __shfl_xor(v, 2, 32));
        v = fmaxf(v, __shfl_xor(v, 4, 32));
        v = fmaxf(v, __shfl_xor(v, 8, 32));
        float mnew = fmaxf(mrow[j], v);
        float corr = __expf(mrow[j] - mnew);
        mrow[j] = mnew;
        float p0 = __expf(s0 - mnew);
        float p1 = __expf(s1 - mnew);
        O0[j] *= corr; O1[j] *= corr; O2[j] *= corr; O3[j] *= corr; O4[j] *= corr;
        // stage P (C-layout -> row-major 16x32) for A-operand reload
        Pw[(j + 8 * hi) * 32 + l]      = (_Float16)p0;
        Pw[(j + 8 * hi) * 32 + 16 + l] = (_Float16)p1;
      }
      v16h pa = loadA(Pw, 32, l, hi, 0);
      O0 = wmma16(pa, loadB(vT + 0 * 16 * SVT, SVT, l, hi, key0), O0);
      O1 = wmma16(pa, loadB(vT + 1 * 16 * SVT, SVT, l, hi, key0), O1);
      O2 = wmma16(pa, loadB(vT + 2 * 16 * SVT, SVT, l, hi, key0), O2);
      O3 = wmma16(pa, loadB(vT + 3 * 16 * SVT, SVT, l, hi, key0), O3);
      O4 = wmma16(pa, onesB, O4);         // row sums ride the matrix pipe
    }
    // normalize by broadcast row sum (lane 0 / lane 16 hold column N=0) & store
    float* ob = out + ((size_t)b * T_SEQ + q0) * H_DIM;
#pragma unroll
    for (int j = 0; j < 8; ++j) {
      float rsum = __shfl(O4[j], hi << 4, 32);
      float inv = 1.0f / rsum;
      int r = j + 8 * hi;
      ob[r * H_DIM +  0 + l] = O0[j] * inv;
      ob[r * H_DIM + 16 + l] = O1[j] * inv;
      ob[r * H_DIM + 32 + l] = O2[j] * inv;
      ob[r * H_DIM + 48 + l] = O3[j] * inv;
    }
  }
}

extern "C" void kernel_launch(void* const* d_in, const int* in_sizes, int n_in,
                              void* d_out, int out_size, void* d_ws, size_t ws_size,
                              hipStream_t stream) {
  const float* x  = (const float*)d_in[0];
  const float* Wk = (const float*)d_in[1];
  const float* bk = (const float*)d_in[2];
  const float* Wq = (const float*)d_in[3];
  const float* bq = (const float*)d_in[4];
  const float* Wv = (const float*)d_in[5];
  const float* bv = (const float*)d_in[6];
  float* outp = (float*)d_out;
  head_attn_kernel<<<2048, 256, 0, stream>>>(x, Wk, bk, Wq, bq, Wv, bv, outp);
}
